// LSTM_Restore_23596550324579
// MI455X (gfx1250) — matrix operand order repaired and checked by reading, not compiled, not run on hardware
//
#include <hip/hip_runtime.h>

// ---------------------------------------------------------------------------
// 2-layer LSTM (H=512, B=256, T=512, D_in=3, D_out=3) for gfx1250 (MI455X).
// bf16 WMMA (v_wmma_f32_16x16x32_bf16) with fp32 accumulate.
// Workgroup = 8 waves sharing one weight column-tile: the 4 gate B-tiles are
// staged once per WG through LDS (async-to-LDS double-buffered via
// global_load_async_to_lds_b128 + s_wait_asynccnt, ds_store fallback
// otherwise), cutting L2 weight traffic 8x. Each wave owns one 16x16
// (batch x hidden) tile, accumulates all 4 gates, then does the pointwise
// LSTM update in registers. Layer 1 fuses the input GEMM (K=1024
// concatenated operand) and the FC head (shfl_xor + f32 atomics).
// ---------------------------------------------------------------------------

typedef __attribute__((ext_vector_type(16))) __bf16 v16bf;
typedef __attribute__((ext_vector_type(8)))  __bf16 v8bf;
typedef __attribute__((ext_vector_type(8)))  float  v8f;
typedef __attribute__((ext_vector_type(4)))  int    v4i_;

#define HD   512      // hidden dim
#define BB   256      // batch
#define TT   512      // time steps
#define G4H  2048     // 4*H

#if defined(__HIP_DEVICE_COMPILE__) && \
    __has_builtin(__builtin_amdgcn_global_load_async_to_lds_b128) && \
    __has_builtin(__builtin_amdgcn_s_wait_asynccnt)
#define USE_ASYNC_LDS 1
// Builtin signature (from hipcc diagnostic): (v4i addrspace(1)*, v4i addrspace(3)*, imm offset, imm cpol)
#define AS1V(p) ((__attribute__((address_space(1))) v4i_*)(p))
#define AS3V(p) ((__attribute__((address_space(3))) v4i_*)(p))
#else
#define USE_ASYNC_LDS 0
#endif

__device__ __forceinline__ float sigmoidf_(float x) {
    return 1.0f / (1.0f + __expf(-x));
}

// A fragment: 16x32 bf16 (row-major A). ISA 7.12.2: lanes 0-15 hold row
// M=lane, K 0..7 (v0-3) and 16..23 (v4-7); lanes 16-31 hold K 8..15 / 24..31.
__device__ __forceinline__ v16bf load_a_frag(const __bf16* __restrict__ A,
                                             int lda, int rowbase, int k0) {
    const int l   = threadIdx.x & 31;
    const int row = rowbase + (l & 15);
    const int kb  = (l >> 4) * 8;
    const __bf16* p = A + (size_t)row * lda + k0 + kb;
    union { v16bf v; struct { v8bf lo, hi; } s; } u;
    u.s.lo = *(const v8bf*)(p);        // K = k0+kb    .. +7
    u.s.hi = *(const v8bf*)(p + 16);   // K = k0+kb+16 .. +23
    return u.v;
}

// ---------------------------------------------------------------------------
// Shared GEMM core: acc[g] += A_tile x W_tile(gate g)^T over K, with the four
// gate B-tiles (32x16 bf16 each, 4 KB total) staged through LDS per WG.
// B frag layout in LDS matches the per-lane fragment: lane l holds 16
// contiguous bf16 (32B) = W[colbase + (l&15)][k0 + (l>>4)*8 ..+7 and +16 ..+23].
// Staging: 256 threads x 16B = 4 KB per K-step, double buffered.
// ---------------------------------------------------------------------------
template<int K>
__device__ __forceinline__ void gates_gemm(const __bf16* __restrict__ A,
                                           const __bf16* __restrict__ W,
                                           int jc, int rowbase,
                                           __bf16* Bs,      // [2*2048] bf16
                                           v8f acc[4])
{
    const int tid  = threadIdx.x;
    const int lane = tid & 31;
    constexpr int NK = K / 32;

    // staging role of this thread: gate sg, fragment lane sl, 16B half sh
    const int sg = tid >> 6;
    const int u  = tid & 63;
    const int sl = u >> 1;
    const int sh = u & 1;
    const __bf16* gsrc = W + (size_t)(sg * HD + jc * 16 + (sl & 15)) * K
                           + (sl >> 4) * 8 + sh * 16;
    __bf16* sdst = Bs + (sg * 512 + sl * 16 + sh * 8);

    auto consume = [&](int kk) {
        const __bf16* buf = Bs + ((kk & 1) ? 2048 : 0);
        v16bf a = load_a_frag(A, K, rowbase, kk * 32);
        #pragma unroll
        for (int g = 0; g < 4; ++g) {
            v16bf b = *(const v16bf*)(buf + g * 512 + lane * 16);
            acc[g] = __builtin_amdgcn_wmma_f32_16x16x32_bf16(
                false, a, false, b, (short)0, acc[g], false, false);
        }
    };

#if USE_ASYNC_LDS
    // Async double-buffered pipeline (ASYNCcnt): loads return done in order,
    // so waiting asynccnt<=1 guarantees the previous stage has landed.
    __builtin_amdgcn_global_load_async_to_lds_b128(AS1V(gsrc), AS3V(sdst), 0, 0);
    #pragma unroll 4
    for (int kk = 0; kk < NK - 1; ++kk) {
        __builtin_amdgcn_global_load_async_to_lds_b128(
            AS1V(gsrc + (size_t)(kk + 1) * 32),
            AS3V(sdst + (((kk + 1) & 1) ? 2048 : 0)), 0, 0);
        __builtin_amdgcn_s_wait_asynccnt(1);
        __syncthreads();
        consume(kk);
        __syncthreads();
    }
    __builtin_amdgcn_s_wait_asynccnt(0);
    __syncthreads();
    consume(NK - 1);
#else
    // Register-staged fallback: global -> VGPR (prefetched) -> ds_store.
    v8bf st = *(const v8bf*)gsrc;
    #pragma unroll 4
    for (int kk = 0; kk < NK - 1; ++kk) {
        *(v8bf*)(sdst + ((kk & 1) ? 2048 : 0)) = st;
        st = *(const v8bf*)(gsrc + (size_t)(kk + 1) * 32);
        __syncthreads();
        consume(kk);
        __syncthreads();
    }
    *(v8bf*)(sdst + (((NK - 1) & 1) ? 2048 : 0)) = st;
    __syncthreads();
    consume(NK - 1);
#endif
}

// ---------------------------------------------------------------------------
// Layer 0 step: gates = hA @ Wh^T + bias + x_t @ Wih^T ; LSTM update.
// grid = (32 col-tiles, 2), block = 256 (8 waves; wave w -> batch tile).
// ---------------------------------------------------------------------------
__global__ __launch_bounds__(256)
void lstm0_step_kernel(const __bf16* __restrict__ hA,      // [B,H]  h0_{t-1}
                       __bf16*       __restrict__ hA_next, // [B,H]  h0_t
                       __bf16*       __restrict__ acat,    // [B,1024] cols 0..511
                       const __bf16* __restrict__ Wh,      // [2048,512] bf16
                       const float*  __restrict__ bias,    // [2048]
                       float*        __restrict__ cbuf,    // [B*H] frag-linear
                       const float*  __restrict__ x,       // [B,T,3]
                       const float*  __restrict__ Wih,     // [2048,3]
                       int t)
{
    __shared__ __attribute__((aligned(32))) __bf16 Bs[2 * 2048];
    const int lane = threadIdx.x & 31;
    const int wv   = threadIdx.x >> 5;
    const int jc   = blockIdx.x;                 // hidden col tile 0..31
    const int ir   = blockIdx.y * 8 + wv;        // batch row tile 0..15
    const int rowbase = ir * 16;
    const int n  = lane & 15;
    const int hi = lane >> 4;

    v8f acc[4];
    #pragma unroll
    for (int g = 0; g < 4; ++g) {
        float bv = bias[g * HD + jc * 16 + n];
        acc[g] = (v8f){bv, bv, bv, bv, bv, bv, bv, bv};
    }

    // x_t contribution (D_in = 3): tiny, scalar FMAs.
    #pragma unroll
    for (int g = 0; g < 4; ++g) {
        const float* wr = Wih + (size_t)(g * HD + jc * 16 + n) * 3;
        const float w0 = wr[0], w1 = wr[1], w2 = wr[2];
        #pragma unroll
        for (int e = 0; e < 8; ++e) {
            const int brow = rowbase + e + 8 * hi;
            const float* xr = x + ((size_t)brow * TT + t) * 3;
            acc[g][e] += xr[0] * w0 + xr[1] * w1 + xr[2] * w2;
        }
    }

    gates_gemm<HD>(hA, Wh, jc, rowbase, Bs, acc);

    // Pointwise LSTM update; c kept in fragment-linear layout (coalesced).
    const int tile = ir * 32 + jc;
    float* cp = cbuf + (size_t)tile * 256 + lane * 8;
    #pragma unroll
    for (int e = 0; e < 8; ++e) {
        const float iv = sigmoidf_(acc[0][e]);
        const float fv = sigmoidf_(acc[1][e]);
        const float gv = tanhf(acc[2][e]);
        const float ov = sigmoidf_(acc[3][e]);
        const float c  = fv * cp[e] + iv * gv;
        cp[e] = c;
        const float h = ov * tanhf(c);
        const int brow = rowbase + e + 8 * hi;
        const int hcol = jc * 16 + n;
        const __bf16 hb = (__bf16)h;
        hA_next[(size_t)brow * HD + hcol] = hb;   // layer-0 recurrence
        acat[(size_t)brow * 1024 + hcol]  = hb;   // layer-1 A operand
    }
}

// ---------------------------------------------------------------------------
// Layer 1 step: gates = [h0_t | h1_{t-1}] @ [W_ih1|W_hh1]^T + bias (K=1024),
// LSTM update, fused FC with atomic accumulate (out pre-seeded with b_fc).
// ---------------------------------------------------------------------------
__global__ __launch_bounds__(256)
void lstm1_step_kernel(const __bf16* __restrict__ acat,      // [B,1024]
                       __bf16*       __restrict__ acat_next, // [B,1024] cols 512..
                       const __bf16* __restrict__ Wc,        // [2048,1024] bf16
                       const float*  __restrict__ bias,      // [2048]
                       float*        __restrict__ cbuf,      // [B*H]
                       const float*  __restrict__ Wfc,       // [3,512]
                       float*        __restrict__ out,       // [B,T,3]
                       int t)
{
    __shared__ __attribute__((aligned(32))) __bf16 Bs[2 * 2048];
    const int lane = threadIdx.x & 31;
    const int wv   = threadIdx.x >> 5;
    const int jc   = blockIdx.x;
    const int ir   = blockIdx.y * 8 + wv;
    const int rowbase = ir * 16;
    const int n  = lane & 15;
    const int hi = lane >> 4;

    v8f acc[4];
    #pragma unroll
    for (int g = 0; g < 4; ++g) {
        float bv = bias[g * HD + jc * 16 + n];
        acc[g] = (v8f){bv, bv, bv, bv, bv, bv, bv, bv};
    }

    gates_gemm<1024>(acat, Wc, jc, rowbase, Bs, acc);

    const int tile = ir * 32 + jc;
    const int hcol = jc * 16 + n;
    float* cp = cbuf + (size_t)tile * 256 + lane * 8;
    const float wf0 = Wfc[0 * HD + hcol];
    const float wf1 = Wfc[1 * HD + hcol];
    const float wf2 = Wfc[2 * HD + hcol];

    #pragma unroll
    for (int e = 0; e < 8; ++e) {
        const float iv = sigmoidf_(acc[0][e]);
        const float fv = sigmoidf_(acc[1][e]);
        const float gv = tanhf(acc[2][e]);
        const float ov = sigmoidf_(acc[3][e]);
        const float c  = fv * cp[e] + iv * gv;
        cp[e] = c;
        const float h = ov * tanhf(c);
        const int brow = rowbase + e + 8 * hi;
        acat_next[(size_t)brow * 1024 + 512 + hcol] = (__bf16)h;

        // Fused FC: reduce h*wfc over the 16 N-lanes of this half-wave.
        float p0 = h * wf0, p1 = h * wf1, p2 = h * wf2;
        #pragma unroll
        for (int m = 1; m < 16; m <<= 1) {
            p0 += __shfl_xor(p0, m, 32);
            p1 += __shfl_xor(p1, m, 32);
            p2 += __shfl_xor(p2, m, 32);
        }
        if (n == 0) {
            float* op = out + ((size_t)brow * TT + t) * 3;
            atomicAdd(op + 0, p0);
            atomicAdd(op + 1, p1);
            atomicAdd(op + 2, p2);
        }
    }
}

// ---------------------------------------------------------------------------
// Init: zero state, fold biases, pre-seed out with b_fc.
// ---------------------------------------------------------------------------
__global__ void init_kernel(__bf16* hA0a, __bf16* hA0b,
                            __bf16* acatA, __bf16* acatB,
                            float* c0, float* c1,
                            float* bias0, float* bias1,
                            const float* b_ih0, const float* b_hh0,
                            const float* b_ih1, const float* b_hh1,
                            float* out, const float* b_fc)
{
    const int i = blockIdx.x * blockDim.x + threadIdx.x;
    if (i < BB * HD)   { hA0a[i] = (__bf16)0.f; hA0b[i] = (__bf16)0.f;
                         c0[i] = 0.f; c1[i] = 0.f; }
    if (i < BB * 1024) { acatA[i] = (__bf16)0.f; acatB[i] = (__bf16)0.f; }
    if (i < G4H)       { bias0[i] = b_ih0[i] + b_hh0[i];
                         bias1[i] = b_ih1[i] + b_hh1[i]; }
    if (i < BB * TT * 3) out[i] = b_fc[i % 3];
}

__global__ void convert_w_kernel(const float* Whh0, const float* Wih1,
                                 const float* Whh1,
                                 __bf16* wh0, __bf16* wcat)
{
    const int i = blockIdx.x * blockDim.x + threadIdx.x;
    if (i < G4H * HD) wh0[i] = (__bf16)Whh0[i];
    if (i < G4H * 1024) {
        const int r = i >> 10, c = i & 1023;
        const float v = (c < HD) ? Wih1[(size_t)r * HD + c]
                                 : Whh1[(size_t)r * HD + (c - HD)];
        wcat[i] = (__bf16)v;
    }
}

// ---------------------------------------------------------------------------
extern "C" void kernel_launch(void* const* d_in, const int* in_sizes, int n_in,
                              void* d_out, int out_size, void* d_ws, size_t ws_size,
                              hipStream_t stream) {
    const float* x     = (const float*)d_in[0];
    const float* W_ih0 = (const float*)d_in[1];
    const float* W_hh0 = (const float*)d_in[2];
    const float* b_ih0 = (const float*)d_in[3];
    const float* b_hh0 = (const float*)d_in[4];
    const float* W_ih1 = (const float*)d_in[5];
    const float* W_hh1 = (const float*)d_in[6];
    const float* b_ih1 = (const float*)d_in[7];
    const float* b_hh1 = (const float*)d_in[8];
    const float* W_fc  = (const float*)d_in[9];
    const float* b_fc  = (const float*)d_in[10];
    float* out = (float*)d_out;

    char* w = (char*)d_ws;
    size_t off = 0;
    auto alloc = [&](size_t bytes) -> void* {
        void* p = (void*)(w + off);
        off += (bytes + 255) & ~(size_t)255;
        return p;
    };
    __bf16* wh0   = (__bf16*)alloc((size_t)G4H * HD   * 2);  // 2 MB
    __bf16* wcat  = (__bf16*)alloc((size_t)G4H * 1024 * 2);  // 4 MB
    float*  bias0 = (float*) alloc((size_t)G4H * 4);
    float*  bias1 = (float*) alloc((size_t)G4H * 4);
    __bf16* hA0a  = (__bf16*)alloc((size_t)BB * HD * 2);
    __bf16* hA0b  = (__bf16*)alloc((size_t)BB * HD * 2);
    __bf16* acatA = (__bf16*)alloc((size_t)BB * 1024 * 2);
    __bf16* acatB = (__bf16*)alloc((size_t)BB * 1024 * 2);
    float*  c0    = (float*) alloc((size_t)BB * HD * 4);
    float*  c1    = (float*) alloc((size_t)BB * HD * 4);

    init_kernel<<<(BB * TT * 3 + 255) / 256, 256, 0, stream>>>(
        hA0a, hA0b, acatA, acatB, c0, c1, bias0, bias1,
        b_ih0, b_hh0, b_ih1, b_hh1, out, b_fc);
    convert_w_kernel<<<(G4H * 1024 + 255) / 256, 256, 0, stream>>>(
        W_hh0, W_ih1, W_hh1, wh0, wcat);

    const dim3 grid(HD / 16, BB / 16 / 8);   // 32 x 2 WGs, 8 waves each
    for (int t = 0; t < TT; ++t) {
        const bool odd = t & 1;
        const __bf16* hin = odd ? hA0b : hA0a;
        __bf16*      hout = odd ? hA0a : hA0b;
        __bf16* acat_cur  = odd ? acatB : acatA;  // h0_t written; h1_{t-1} read
        __bf16* acat_next = odd ? acatA : acatB;  // h1_t written

        lstm0_step_kernel<<<grid, 256, 0, stream>>>(
            hin, hout, acat_cur, wh0, bias0, c0, x, W_ih0, t);
        lstm1_step_kernel<<<grid, 256, 0, stream>>>(
            acat_cur, acat_next, wcat, bias1, c1, W_fc, out, t);
    }
}
